// Attention_org_cross_35261681500334
// MI455X (gfx1250) — matrix operand order repaired and checked by reading, not compiled, not run on hardware
//
#include <hip/hip_runtime.h>

// CDNA5 / gfx1250, wave32. All GEMMs use v_wmma_f32_16x16x32_bf16 (f32 accum).
typedef __attribute__((ext_vector_type(16))) __bf16 v16bf;
typedef __attribute__((ext_vector_type(8)))  float  v8f;

#define WMMA_BF16(A, B, C) \
  __builtin_amdgcn_wmma_f32_16x16x32_bf16(false, (A), false, (B), (short)0, (C), false, false)

// dir 0 = optical queries (emb1..4, attend emb_alld K/V via wkd/wvd, out -> ch [0,960))
// dir 1 = DSM queries    (embd1..4, attend emb_all  K/V via wk/wv,  out -> ch [960,1920))
struct Ptrs {
  const float* q[2][4];     // per-dir query embeddings per scale
  const float* kv[2];       // kv[0]=emb_alld, kv[1]=emb_all
  const float* wq[2][4];    // (H,cq,cq)
  const float* wout[2][4];  // (c,c)
  const float* wk[2];       // kv-side key weight for that dir's attention
  const float* wv[2];
};

__device__ __forceinline__ int scale_of(int c) { return (c >= 112) ? 3 : (c >= 48) ? 2 : (c >= 16) ? 1 : 0; }
__device__ __forceinline__ int joffs(int s) { return 16 * ((1 << s) - 1); }   // 0,16,48,112
__device__ __forceinline__ int coffs(int s) { return 64 * ((1 << s) - 1); }   // 0,64,192,448

// ---------------------------------------------------------------------------
// K0: pre-pack permuted emb_all / emb_alld into bf16 WMMA A-fragments.
// Afrag[dir][b][ntile(256)][chunk(30)][lane(32)][e(16)]  (bf16)
// A element (lane,e): n = ntile*16 + (lane&15), q = chunk*32 + ((lane&16)>>1) + e + (e&8)
// ---------------------------------------------------------------------------
__global__ __launch_bounds__(256) void k0_afrag(Ptrs p, unsigned short* AfragU) {
  __bf16* Afrag = (__bf16*)AfragU;
  int bid = blockIdx.x;
  int chunk = bid % 30; bid /= 30;
  int t = bid % 256;    bid /= 256;
  int b = bid % 4;      bid /= 4;
  int dir = bid;
  int tid = threadIdx.x;
  int lane = tid >> 3;
  int e0 = (tid & 7) * 2;
  const float* src = p.kv[dir] + ((size_t)b * 4096 + t * 16 + (lane & 15)) * 960;
  size_t ob = ((((size_t)(dir * 4 + b)) * 256 + t) * 30 + chunk) * 512 + (size_t)lane * 16;
  int kaOff = (lane & 16) >> 1;
#pragma unroll
  for (int u = 0; u < 2; ++u) {
    int e = e0 + u;
    int q = chunk * 32 + kaOff + e + (e & 8);
    int h = q / 240;
    int j = q - h * 240;
    int s = scale_of(j);
    int ch = coffs(s) + h * (16 << s) + (j - joffs(s));
    Afrag[ob + e] = (__bf16)src[ch];
  }
}

// ---------------------------------------------------------------------------
// K1: G[dir][b][h] (240x240) = Xall^T (240x4096) * KVQ (4096x240), contract N.
// 3x3 register-blocked: each wave computes a 48x48 block (9 tiles); 6 fragment
// builds feed 9 WMMAs per 32-deep K step. Loads stay 64B-coalesced.
// ---------------------------------------------------------------------------
__global__ __launch_bounds__(128) void k1_g(Ptrs p, float* G) {
  int wave = blockIdx.x * 4 + (threadIdx.x >> 5);  // 2*4*4*5*5 = 800 waves
  int lane = threadIdx.x & 31;
  int gj = wave % 5; wave /= 5;
  int gi = wave % 5; wave /= 5;
  int h = wave % 4;  wave /= 4;
  int b = wave % 4;  wave /= 4;
  int dir = wave;
  int m = lane & 15;

  const float* baseA[3];
  int ldA[3];
  const float* baseB[3];
#pragma unroll
  for (int u = 0; u < 3; ++u) {
    int c = (gi * 3 + u) * 16 + m;
    int si = scale_of(c);
    int cq = 16 << si, ld = 64 << si;
    ldA[u] = ld;
    baseA[u] = p.q[dir][si] + ((size_t)b * 4096) * ld + h * cq + (c - joffs(si));
    int j = (gj * 3 + u) * 16 + m;
    int sj = scale_of(j);
    baseB[u] = p.kv[dir] + ((size_t)b * 4096) * 960 + coffs(sj) + h * (16 << sj) + (j - joffs(sj));
  }

  int kaOff = (lane & 16) >> 1, kbOff = lane & 16;
  v8f acc[3][3];
#pragma unroll
  for (int u = 0; u < 3; ++u)
#pragma unroll
    for (int v = 0; v < 3; ++v) acc[u][v] = {};

  for (int k0 = 0; k0 < 4096; k0 += 32) {
    v16bf a[3], bb[3];
#pragma unroll
    for (int u = 0; u < 3; ++u) {
#pragma unroll
      for (int e = 0; e < 16; ++e) {
        int ka = k0 + kaOff + e + (e & 8);
        int kb = k0 + kaOff + e + (e & 8);
        a[u][e]  = (__bf16)baseA[u][(size_t)ka * ldA[u]];
        bb[u][e] = (__bf16)baseB[u][(size_t)kb * 960];
      }
    }
#pragma unroll
    for (int u = 0; u < 3; ++u)
#pragma unroll
      for (int v = 0; v < 3; ++v) acc[u][v] = WMMA_BF16(a[u], bb[v], acc[u][v]);
  }

  float* g = G + ((((size_t)dir * 4 + b) * 4) + h) * 57600;
#pragma unroll
  for (int u = 0; u < 3; ++u)
#pragma unroll
    for (int v = 0; v < 3; ++v)
#pragma unroll
      for (int r = 0; r < 8; ++r)
        g[(size_t)((gi * 3 + u) * 16 + r + ((lane & 16) >> 1)) * 240 + (gj * 3 + v) * 16 + m] =
            acc[u][v][r];
}

// ---------------------------------------------------------------------------
// K2a: T1[d,k] = sum_j G[d,j] * wk[k,j]   (K padded 240 -> 256 with guards)
// ---------------------------------------------------------------------------
__global__ __launch_bounds__(128) void k2a_t1(Ptrs p, const float* G, float* T1) {
  int wave = blockIdx.x * 4 + (threadIdx.x >> 5);
  int lane = threadIdx.x & 31, m = lane & 15;
  int tj = wave % 15; wave /= 15;
  int ti = wave % 15; wave /= 15;
  int h = wave % 4;   wave /= 4;
  int b = wave % 4;   wave /= 4;
  int dir = wave;
  size_t bh = (((size_t)dir * 4 + b) * 4) + h;
  const float* baseA = G + bh * 57600 + (size_t)(ti * 16 + m) * 240;
  const float* baseB = p.wk[dir] + (size_t)(tj * 16 + m) * 240;
  int kaOff = (lane & 16) >> 1, kbOff = lane & 16;
  v8f acc = {};
  for (int k0 = 0; k0 < 256; k0 += 32) {
    v16bf a, bb;
#pragma unroll
    for (int e = 0; e < 16; ++e) {
      int ja = k0 + kaOff + e + (e & 8);
      int jb = k0 + kaOff + e + (e & 8);
      a[e]  = (ja < 240) ? (__bf16)baseA[ja] : (__bf16)0.0f;
      bb[e] = (jb < 240) ? (__bf16)baseB[jb] : (__bf16)0.0f;
    }
    acc = WMMA_BF16(a, bb, acc);
  }
  float* t1 = T1 + bh * 57600;
#pragma unroll
  for (int r = 0; r < 8; ++r)
    t1[(size_t)(ti * 16 + r + ((lane & 16) >> 1)) * 240 + tj * 16 + m] = acc[r];
}

// ---------------------------------------------------------------------------
// K2b: scores[d,k] = (1/sqrt(960)) * sum_c wq[h,d,c] * T1[joffs+c, k]
// ---------------------------------------------------------------------------
__global__ __launch_bounds__(128) void k2b_scores(Ptrs p, const float* T1, float* S) {
  int wave = blockIdx.x * 4 + (threadIdx.x >> 5);
  int lane = threadIdx.x & 31, m = lane & 15;
  int tj = wave % 15; wave /= 15;
  int ti = wave % 15; wave /= 15;
  int h = wave % 4;   wave /= 4;
  int b = wave % 4;   wave /= 4;
  int dir = wave;
  size_t bh = (((size_t)dir * 4 + b) * 4) + h;
  int d = ti * 16 + m;
  int si = scale_of(d);
  int cq = 16 << si;
  const float* baseA = p.wq[dir][si] + (size_t)(h * cq + (d - joffs(si))) * cq;
  const float* baseB = T1 + bh * 57600 + (size_t)joffs(si) * 240 + tj * 16 + m;
  int kaOff = (lane & 16) >> 1, kbOff = lane & 16;
  int Kt = (cq < 32) ? 32 : cq;
  v8f acc = {};
  for (int k0 = 0; k0 < Kt; k0 += 32) {
    v16bf a, bb;
#pragma unroll
    for (int e = 0; e < 16; ++e) {
      int ca = k0 + kaOff + e + (e & 8);
      int cb = k0 + kaOff + e + (e & 8);
      a[e]  = (ca < cq) ? (__bf16)baseA[ca] : (__bf16)0.0f;
      bb[e] = (cb < cq) ? (__bf16)baseB[(size_t)cb * 240] : (__bf16)0.0f;
    }
    acc = WMMA_BF16(a, bb, acc);
  }
  float* s = S + bh * 57600;
  const float sc = 0.03227486121839514f;  // 1/sqrt(960)
#pragma unroll
  for (int r = 0; r < 8; ++r)
    s[(size_t)(ti * 16 + r + ((lane & 16) >> 1)) * 240 + tj * 16 + m] = acc[r] * sc;
}

// ---------------------------------------------------------------------------
// K2c: per (dir,b,h,scale) instance-norm over (cq,240) then row softmax (k).
// ---------------------------------------------------------------------------
__global__ __launch_bounds__(256) void k2c_softmax(float* S) {
  int bid = blockIdx.x;              // 2*4*4*4 = 128
  int si = bid & 3;
  int h = (bid >> 2) & 3;
  int b = (bid >> 4) & 3;
  int dir = bid >> 6;
  int cq = 16 << si;
  float* blk = S + ((((size_t)dir * 4 + b) * 4) + h) * 57600 + (size_t)joffs(si) * 240;
  int n = cq * 240;
  int tid = threadIdx.x;
  float s = 0.f, s2 = 0.f;
  for (int i = tid; i < n; i += 256) { float v = blk[i]; s += v; s2 += v * v; }
  __shared__ float r1[256], r2[256];
  r1[tid] = s; r2[tid] = s2;
  __syncthreads();
  for (int o = 128; o > 0; o >>= 1) {
    if (tid < o) { r1[tid] += r1[tid + o]; r2[tid] += r2[tid + o]; }
    __syncthreads();
  }
  float mean = r1[0] / (float)n;
  float var = r2[0] / (float)n - mean * mean;
  float inv = rsqrtf(var + 1e-5f);
  if (tid < cq) {
    float* row = blk + (size_t)tid * 240;
    float mx = -3.0e38f;
    for (int k = 0; k < 240; ++k) mx = fmaxf(mx, (row[k] - mean) * inv);
    float sum = 0.f;
    for (int k = 0; k < 240; ++k) { float v = __expf((row[k] - mean) * inv - mx); row[k] = v; sum += v; }
    float is = 1.0f / sum;
    for (int k = 0; k < 240; ++k) row[k] *= is;
  }
}

// ---------------------------------------------------------------------------
// K2d: P2[c,j] = sum_k probs[c,k] * wv[k,j]
// ---------------------------------------------------------------------------
__global__ __launch_bounds__(128) void k2d_p2(Ptrs p, const float* S, float* P2) {
  int wave = blockIdx.x * 4 + (threadIdx.x >> 5);
  int lane = threadIdx.x & 31, m = lane & 15;
  int tj = wave % 15; wave /= 15;
  int ti = wave % 15; wave /= 15;
  int h = wave % 4;   wave /= 4;
  int b = wave % 4;   wave /= 4;
  int dir = wave;
  size_t bh = (((size_t)dir * 4 + b) * 4) + h;
  const float* baseA = S + bh * 57600 + (size_t)(ti * 16 + m) * 240;
  const float* baseB = p.wv[dir] + (tj * 16 + m);
  int kaOff = (lane & 16) >> 1, kbOff = lane & 16;
  v8f acc = {};
  for (int k0 = 0; k0 < 256; k0 += 32) {
    v16bf a, bb;
#pragma unroll
    for (int e = 0; e < 16; ++e) {
      int ka = k0 + kaOff + e + (e & 8);
      int kb = k0 + kaOff + e + (e & 8);
      a[e]  = (ka < 240) ? (__bf16)baseA[ka] : (__bf16)0.0f;
      bb[e] = (kb < 240) ? (__bf16)baseB[(size_t)kb * 240] : (__bf16)0.0f;
    }
    acc = WMMA_BF16(a, bb, acc);
  }
  float* p2 = P2 + bh * 57600;
#pragma unroll
  for (int r = 0; r < 8; ++r)
    p2[(size_t)(ti * 16 + r + ((lane & 16) >> 1)) * 240 + tj * 16 + m] = acc[r];
}

// ---------------------------------------------------------------------------
// K2e: WfoldT[q=(h*240+j), m] = sum_c wout[m_local, h*cq+c] * P2[joffs+c, j]
// stored directly as bf16 B-fragments for K3:
// Bfrag[dir][b][mtile(60)][chunk(30)][lane][e], B elem (lane,e): K=chunk*32+((lane&16)>>1)+e+(e&8), N=lane&15
// ---------------------------------------------------------------------------
__global__ __launch_bounds__(128) void k2e_bfrag(Ptrs p, const float* P2, unsigned short* BfragU) {
  __bf16* Bfrag = (__bf16*)BfragU;
  int wave = blockIdx.x * 4 + (threadIdx.x >> 5);   // 2*4*60*60 = 28800
  int lane = threadIdx.x & 31, m = lane & 15;
  int mt = wave % 60; wave /= 60;
  int qt = wave % 60; wave /= 60;
  int b = wave % 4;   wave /= 4;
  int dir = wave;
  int h = qt / 15;
  int j = (qt % 15) * 16 + m;                      // per-lane A row -> j
  int m0 = mt * 16;
  int sm = (m0 >= 448) ? 3 : (m0 >= 192) ? 2 : (m0 >= 64) ? 1 : 0;
  int cq = 16 << sm, ci = 64 << sm;
  const float* baseA = P2 + ((((size_t)dir * 4 + b) * 4) + h) * 57600 + (size_t)joffs(sm) * 240 + j;
  const float* baseB = p.wout[dir][sm] + (size_t)(m0 + m - coffs(sm)) * ci + h * cq;
  int kaOff = (lane & 16) >> 1, kbOff = lane & 16;
  int Kt = (cq < 32) ? 32 : cq;
  v8f acc = {};
  for (int k0 = 0; k0 < Kt; k0 += 32) {
    v16bf a, bb;
#pragma unroll
    for (int e = 0; e < 16; ++e) {
      int ca = k0 + kaOff + e + (e & 8);
      int cb = k0 + kaOff + e + (e & 8);
      a[e]  = (ca < cq) ? (__bf16)baseA[(size_t)ca * 240] : (__bf16)0.0f;
      bb[e] = (cb < cq) ? (__bf16)baseB[cb] : (__bf16)0.0f;
    }
    acc = WMMA_BF16(a, bb, acc);
  }
  size_t mbase = (((size_t)(dir * 4 + b)) * 60 + mt) * 30;
#pragma unroll
  for (int r = 0; r < 8; ++r) {
    int q = qt * 16 + r + ((lane & 16) >> 1);
    int chunk = q >> 5, kl = q & 31;
    int lane2 = ((kl & 8) << 1) | m;
    int e2 = (kl & 7) | ((kl & 16) >> 1);
    Bfrag[(mbase + chunk) * 512 + (size_t)lane2 * 16 + e2] = (__bf16)acc[r];
  }
}

// ---------------------------------------------------------------------------
// K3: out[b, n, ch] = sum_q Afrag * Bfrag  — (4096 x 960) x (960 x 960) per (b,dir).
// 4x4 register-blocked: each wave computes a 64x64 output block. Per 32-deep
// K step: 8 v16bf fragment loads (16x b128) feed 16 WMMAs -> 1 b128 per WMMA.
// m-tile groups of 4 never straddle the dir boundary (60 % 4 == 0).
// ---------------------------------------------------------------------------
__global__ __launch_bounds__(128) void k3_out(const unsigned short* AfragU, const unsigned short* BfragU,
                                              float* out) {
  const __bf16* Afrag = (const __bf16*)AfragU;
  const __bf16* Bfrag = (const __bf16*)BfragU;
  int wave = blockIdx.x * 4 + (threadIdx.x >> 5);  // 4 * 64 * 30 = 7680 waves
  int lane = threadIdx.x & 31;
  int cg = wave % 30; wave /= 30;                  // group of 4 m-tiles
  int ng = wave % 64; wave /= 64;                  // group of 4 n-tiles
  int b = wave;
  int dir = (cg >= 15) ? 1 : 0;
  int mt0 = (cg - dir * 15) * 4;
  int t0 = ng * 4;

  const __bf16* af[4];
  const __bf16* bf[4];
#pragma unroll
  for (int u = 0; u < 4; ++u) {
    af[u] = Afrag + ((((size_t)(dir * 4 + b)) * 256 + (t0 + u)) * 30) * 512 + (size_t)lane * 16;
    bf[u] = Bfrag + ((((size_t)(dir * 4 + b)) * 60 + (mt0 + u)) * 30) * 512 + (size_t)lane * 16;
  }

  v8f acc[4][4];
#pragma unroll
  for (int u = 0; u < 4; ++u)
#pragma unroll
    for (int v = 0; v < 4; ++v) acc[u][v] = {};

  for (int ch = 0; ch < 30; ++ch) {
    v16bf a[4], bb[4];
#pragma unroll
    for (int u = 0; u < 4; ++u) {
      a[u]  = *(const v16bf*)(af[u] + (size_t)ch * 512);
      bb[u] = *(const v16bf*)(bf[u] + (size_t)ch * 512);
    }
#pragma unroll
    for (int u = 0; u < 4; ++u)
#pragma unroll
      for (int v = 0; v < 4; ++v) acc[u][v] = WMMA_BF16(a[u], bb[v], acc[u][v]);
  }

#pragma unroll
  for (int u = 0; u < 4; ++u) {
    int n0 = (t0 + u) * 16;
#pragma unroll
    for (int v = 0; v < 4; ++v) {
      int c0 = dir * 960 + (mt0 + v) * 16;
      float* o = out + ((size_t)b * 4096 + n0) * 1920 + c0 + (lane & 15);
#pragma unroll
      for (int r = 0; r < 8; ++r)
        o[(size_t)(r + ((lane & 16) >> 1)) * 1920] = acc[u][v][r];
    }
  }
}

// ---------------------------------------------------------------------------
extern "C" void kernel_launch(void* const* d_in, const int* in_sizes, int n_in,
                              void* d_out, int out_size, void* d_ws, size_t ws_size,
                              hipStream_t stream) {
  (void)in_sizes; (void)n_in; (void)out_size; (void)ws_size;
  Ptrs p;
  // dict order: emb1, embd1, emb2, embd2, emb3, embd3, emb4, embd4, emb_all, emb_alld,
  //             (wq_i, wqd_i, wout_i, woutd_i) x4, wk, wv, wkd, wvd
  p.q[0][0] = (const float*)d_in[0]; p.q[1][0] = (const float*)d_in[1];
  p.q[0][1] = (const float*)d_in[2]; p.q[1][1] = (const float*)d_in[3];
  p.q[0][2] = (const float*)d_in[4]; p.q[1][2] = (const float*)d_in[5];
  p.q[0][3] = (const float*)d_in[6]; p.q[1][3] = (const float*)d_in[7];
  p.kv[1] = (const float*)d_in[8];   // emb_all  (dsm queries attend K/V)
  p.kv[0] = (const float*)d_in[9];   // emb_alld (optical queries attend Kd/Vd)
  for (int i = 0; i < 4; ++i) {
    p.wq[0][i]   = (const float*)d_in[10 + 4 * i];
    p.wq[1][i]   = (const float*)d_in[11 + 4 * i];
    p.wout[0][i] = (const float*)d_in[12 + 4 * i];
    p.wout[1][i] = (const float*)d_in[13 + 4 * i];
  }
  p.wk[1] = (const float*)d_in[26];  // wk  (dsm path)
  p.wv[1] = (const float*)d_in[27];  // wv
  p.wk[0] = (const float*)d_in[28];  // wkd (optical path)
  p.wv[0] = (const float*)d_in[29];  // wvd

  char* ws = (char*)d_ws;
  const size_t SZ = 2ull * 4 * 4 * 240 * 240 * 4;  // 7,372,800 B per f32 buffer
  float* G  = (float*)(ws + 0 * SZ);
  float* T1 = (float*)(ws + 1 * SZ);
  float* S  = (float*)(ws + 2 * SZ);
  float* P2 = (float*)(ws + 3 * SZ);
  unsigned short* Bf = (unsigned short*)(ws + 4 * SZ);                 // 14,745,600 B
  unsigned short* Af = (unsigned short*)(ws + 4 * SZ + 14745600ull);   // 62,914,560 B
  float* out = (float*)d_out;

  k0_afrag   <<<61440, 256, 0, stream>>>(p, Af);
  k1_g       <<<200, 128, 0, stream>>>(p, G);     // 800 waves, 3x3 blocked
  k2a_t1     <<<1800, 128, 0, stream>>>(p, G, T1);
  k2b_scores <<<1800, 128, 0, stream>>>(p, T1, S);
  k2c_softmax<<<128, 256, 0, stream>>>(S);
  k2d_p2     <<<1800, 128, 0, stream>>>(p, S, P2);
  k2e_bfrag  <<<7200, 128, 0, stream>>>(p, P2, Bf);
  k3_out     <<<1920, 128, 0, stream>>>(Af, Bf, out);  // 7680 waves, 4x4 blocked
}
